// PosOnlyModel_21354577395792
// MI455X (gfx1250) — matrix operand order repaired and checked by reading, not compiled, not run on hardware
//
#include <hip/hip_runtime.h>

#define H 256
#define NG 1024
#define MT 32            // batch rows per workgroup (2 row-tiles of 16)
#define MW 15            // encoder window
#define HW 25            // decoder window
#define B_TOTAL 8192
#define AS 544           // abuf K-stride: [0,256)=h0 [256,512)=h1 [512,544)=x (padded)
#define K0W 288          // layer0 packed weight K (256 + 32)
#define K1W 512          // layer1 packed weight K

typedef __attribute__((ext_vector_type(8)))  __bf16 v8bf;
typedef __attribute__((ext_vector_type(16))) __bf16 v16bf;
typedef __attribute__((ext_vector_type(8)))  float  v8f;

__device__ __forceinline__ unsigned short f2bf(float f) {
    unsigned int u = __float_as_uint(f);
    u += 0x7fffu + ((u >> 16) & 1u);          // round-to-nearest-even
    return (unsigned short)(u >> 16);
}
__device__ __forceinline__ float bf2f(unsigned short b) {
    return __uint_as_float(((unsigned int)b) << 16);
}
__device__ __forceinline__ float sigm(float x) { return 1.0f / (1.0f + __expf(-x)); }

// ---------------------------------------------------------------------------
// Dual-tile GEMM: each B fragment (loaded once from L2) feeds TWO WMMAs,
// one per 16-row tile -> halves weight traffic per FLOP vs single-tile.
// A in LDS (bf16 row-major, stride AS); W in global (1024 x w_stride, N x K).
// Fragment layouts per CDNA5 ISA 7.12.2 (wave32, 16-bit A 16x32, B 32x16).
// ---------------------------------------------------------------------------
__device__ __forceinline__ void gemm_bf16_x2(v8f acc0[8], v8f acc1[8],
        const unsigned short* __restrict__ aLDS, int a_k0,
        const unsigned short* __restrict__ W,   int w_stride, int w_k0,
        int nkt, int lane, int wave)
{
    const int mr = lane & 15;        // A row / B column within tile
    const int kh = lane >> 4;        // K-half select
    const int n0 = wave * 128;
#pragma unroll 1
    for (int kt = 0; kt < nkt; ++kt) {
        const unsigned short* ap0 = aLDS + mr * AS + a_k0 + kt * 32;        // rows 0..15
        const unsigned short* ap1 = aLDS + (16 + mr) * AS + a_k0 + kt * 32; // rows 16..31
        v8bf a0lo = *(const v8bf*)(ap0 + kh * 8);
        v8bf a0hi = *(const v8bf*)(ap0 + 16 + kh * 8);
        v8bf a1lo = *(const v8bf*)(ap1 + kh * 8);
        v8bf a1hi = *(const v8bf*)(ap1 + 16 + kh * 8);
        v16bf a0, a1;
#pragma unroll
        for (int i = 0; i < 8; ++i) {
            a0[i] = a0lo[i]; a0[8 + i] = a0hi[i];
            a1[i] = a1lo[i]; a1[8 + i] = a1hi[i];
        }
#pragma unroll
        for (int nt = 0; nt < 8; ++nt) {
            const unsigned short* wp =
                W + (n0 + nt * 16 + mr) * w_stride + w_k0 + kt * 32 + kh * 8;
            v8bf blo = *(const v8bf*)(wp);
            v8bf bhi = *(const v8bf*)(wp + 16);
            v16bf b;
#pragma unroll
            for (int i = 0; i < 8; ++i) { b[i] = blo[i]; b[8 + i] = bhi[i]; }
            acc0[nt] = __builtin_amdgcn_wmma_f32_16x16x32_bf16(
                false, a0, false, b, (short)0, acc0[nt], false, false);
            acc1[nt] = __builtin_amdgcn_wmma_f32_16x16x32_bf16(
                false, a1, false, b, (short)0, acc1[nt], false, false);
        }
    }
}

// ---------------------------------------------------------------------------
// LSTM pointwise update directly from WMMA accumulators (one 16-row tile).
// Column permutation: n' = wave*128 + gate*32 + st*16 + col -> acc[gate*2+st].
// ---------------------------------------------------------------------------
__device__ __forceinline__ void lstm_epilogue(const v8f acc[8],
        const float* __restrict__ biasP, float* creg,
        unsigned short* __restrict__ hb,  // abuf + h offset, row base pre-applied
        int lane, int wave)
{
    const int col   = lane & 15;
    const int rbase = (lane >> 4) * 8;
#pragma unroll
    for (int st = 0; st < 2; ++st) {
        const int hcol = wave * 32 + st * 16 + col;
        const float bi = biasP[wave * 128 + 0 * 32 + st * 16 + col];
        const float bf = biasP[wave * 128 + 1 * 32 + st * 16 + col];
        const float bg = biasP[wave * 128 + 2 * 32 + st * 16 + col];
        const float bo = biasP[wave * 128 + 3 * 32 + st * 16 + col];
#pragma unroll
        for (int r = 0; r < 8; ++r) {
            const int m = rbase + r;
            const float gi = acc[0 + st][r] + bi;
            const float gf = acc[2 + st][r] + bf;
            const float gg = acc[4 + st][r] + bg;
            const float go = acc[6 + st][r] + bo;
            const float c  = sigm(gf) * creg[st * 8 + r] + sigm(gi) * tanhf(gg);
            const float h  = sigm(go) * tanhf(c);
            creg[st * 8 + r] = c;
            hb[m * AS + hcol] = f2bf(h);
        }
    }
}

// ---------------------------------------------------------------------------
// Weight conversion / packing kernel (f32 -> permuted bf16 images in d_ws).
// ---------------------------------------------------------------------------
struct ConvParams {
    const float* Whh0[2];   // enc, dec (1024x256)
    const float* Wih0[2];   // (1024x3)
    const float* Wih1[2];   // (1024x256)
    const float* Whh1[2];   // (1024x256)
    const float* bih0[2]; const float* bhh0[2];
    const float* bih1[2]; const float* bhh1[2];
    unsigned short* wq;     // packed weights
    float* bias;            // 4 x 1024 fused, permuted biases
};

#define E0 (NG * K0W)       // 294912 elems per layer0 image
#define E1 (NG * K1W)       // 524288 elems per layer1 image
#define WQ_TOTAL (2 * (E0 + E1))

__global__ __launch_bounds__(256) void pack_weights_kernel(ConvParams cp) {
    const int idx = blockIdx.x * 256 + threadIdx.x;
    if (idx < WQ_TOTAL) {
        int rem = idx, which;                 // 0=encL0 1=encL1 2=decL0 3=decL1
        if      (rem < E0)           { which = 0; }
        else if (rem < E0 + E1)      { which = 1; rem -= E0; }
        else if (rem < 2*E0 + E1)    { which = 2; rem -= E0 + E1; }
        else                         { which = 3; rem -= 2*E0 + E1; }
        const int ed = which >> 1;            // 0=enc 1=dec
        const int l1 = which & 1;             // 0=layer0 1=layer1
        const int kw = l1 ? K1W : K0W;
        const int np = rem / kw;              // permuted gate column
        const int k  = rem % kw;
        const int w  = np >> 7;
        const int q  = np & 127;
        const int g  = q >> 5;
        const int no = g * 256 + w * 32 + (q & 31);   // original gate row
        float v = 0.0f;
        if (!l1) {
            if (k < 256)           v = cp.Whh0[ed][no * 256 + k];
            else if (k - 256 < 3)  v = cp.Wih0[ed][no * 3 + (k - 256)];
        } else {
            if (k < 256)           v = cp.Wih1[ed][no * 256 + k];
            else                   v = cp.Whh1[ed][no * 256 + (k - 256)];
        }
        cp.wq[idx] = f2bf(v);
    }
    const int bidx = idx - WQ_TOTAL;
    if (bidx >= 0 && bidx < 4 * NG) {         // 0=encB0 1=encB1 2=decB0 3=decB1
        const int which = bidx / NG;
        const int np    = bidx % NG;
        const int ed = which >> 1, l1 = which & 1;
        const int w = np >> 7, q = np & 127, g = q >> 5;
        const int no = g * 256 + w * 32 + (q & 31);
        const float b = l1 ? (cp.bih1[ed][no] + cp.bhh1[ed][no])
                           : (cp.bih0[ed][no] + cp.bhh0[ed][no]);
        cp.bias[bidx] = b;
    }
}

// ---------------------------------------------------------------------------
// Main persistent-state LSTM kernel: 1 block = 32 batch rows, all 40 steps.
// ---------------------------------------------------------------------------
struct Params {
    const float* enc_in;            // (B,15,3)
    const float* dec_in;            // (B,1,3)
    const float* fc_W;              // (3,256)
    const float* fc_b;              // (3)
    const unsigned short* wq;       // packed bf16 weights
    const float* bias;              // 4x1024 fused biases
    float* out;                     // (B,25,3)
};

__global__ __launch_bounds__(256, 1) void lstm_seq_kernel(Params p) {
    const int tid  = threadIdx.x;
    const int lane = tid & 31;
    const int wave = tid >> 5;
    const int b0   = blockIdx.x * MT;

    __shared__ unsigned short abuf[MT * AS];  // bf16 [h0 | h1 | x], 32 rows
    __shared__ float xpos[MT * 4];
    __shared__ float dbuf[MT * 4];

    const unsigned short* encW0 = p.wq;
    const unsigned short* encW1 = p.wq + E0;
    const unsigned short* decW0 = p.wq + E0 + E1;
    const unsigned short* decW1 = p.wq + 2 * E0 + E1;
    const float* encB0 = p.bias + 0 * NG;
    const float* encB1 = p.bias + 1 * NG;
    const float* decB0 = p.bias + 2 * NG;
    const float* decB1 = p.bias + 3 * NG;

    float c0r[32], c1r[32];                   // c-state: [tile0 16 | tile1 16]
#pragma unroll
    for (int j = 0; j < 32; ++j) { c0r[j] = 0.0f; c1r[j] = 0.0f; }
    for (int i = tid; i < MT * AS; i += 256) abuf[i] = 0;   // zero h0,h1,x(+pad)
    __syncthreads();

    // ------------------------- encoder -------------------------
    for (int s = 0; s < MW; ++s) {
        if (tid < MT * 3) {
            const int m = tid / 3, j = tid % 3;
            abuf[m * AS + 512 + j] = f2bf(p.enc_in[(b0 + m) * (MW * 3) + s * 3 + j]);
        }
        __syncthreads();
        {   // layer 0: [h0 | x] @ [Whh0 | Wih0]^T
            v8f acc0[8], acc1[8];
#pragma unroll
            for (int nt = 0; nt < 8; ++nt) { acc0[nt] = (v8f)(0.0f); acc1[nt] = (v8f)(0.0f); }
            gemm_bf16_x2(acc0, acc1, abuf, 0,   encW0, K0W, 0,   8, lane, wave);
            gemm_bf16_x2(acc0, acc1, abuf, 512, encW0, K0W, 256, 1, lane, wave);
            __syncthreads();                       // all waves done reading h0
            lstm_epilogue(acc0, encB0, c0r,      abuf,           lane, wave);
            lstm_epilogue(acc1, encB0, c0r + 16, abuf + 16 * AS, lane, wave);
        }
        __syncthreads();
        {   // layer 1: [h0 | h1] @ [Wih1 | Whh1]^T
            v8f acc0[8], acc1[8];
#pragma unroll
            for (int nt = 0; nt < 8; ++nt) { acc0[nt] = (v8f)(0.0f); acc1[nt] = (v8f)(0.0f); }
            gemm_bf16_x2(acc0, acc1, abuf, 0, encW1, K1W, 0, 16, lane, wave);
            __syncthreads();
            lstm_epilogue(acc0, encB1, c1r,      abuf + 256,           lane, wave);
            lstm_epilogue(acc1, encB1, c1r + 16, abuf + 16 * AS + 256, lane, wave);
        }
        __syncthreads();
    }

    // decoder initial position
    if (tid < MT * 3) {
        const int m = tid / 3, j = tid % 3;
        const float v = p.dec_in[(b0 + m) * 3 + j];
        xpos[m * 4 + j] = v;
        abuf[m * AS + 512 + j] = f2bf(v);
    }
    __syncthreads();

    // ------------------------- decoder -------------------------
    for (int s = 0; s < HW; ++s) {
        {   // layer 0
            v8f acc0[8], acc1[8];
#pragma unroll
            for (int nt = 0; nt < 8; ++nt) { acc0[nt] = (v8f)(0.0f); acc1[nt] = (v8f)(0.0f); }
            gemm_bf16_x2(acc0, acc1, abuf, 0,   decW0, K0W, 0,   8, lane, wave);
            gemm_bf16_x2(acc0, acc1, abuf, 512, decW0, K0W, 256, 1, lane, wave);
            __syncthreads();
            lstm_epilogue(acc0, decB0, c0r,      abuf,           lane, wave);
            lstm_epilogue(acc1, decB0, c0r + 16, abuf + 16 * AS, lane, wave);
        }
        __syncthreads();
        {   // layer 1
            v8f acc0[8], acc1[8];
#pragma unroll
            for (int nt = 0; nt < 8; ++nt) { acc0[nt] = (v8f)(0.0f); acc1[nt] = (v8f)(0.0f); }
            gemm_bf16_x2(acc0, acc1, abuf, 0, decW1, K1W, 0, 16, lane, wave);
            __syncthreads();
            lstm_epilogue(acc0, decB1, c1r,      abuf + 256,           lane, wave);
            lstm_epilogue(acc1, decB1, c1r + 16, abuf + 16 * AS + 256, lane, wave);
        }
        __syncthreads();
        // FC head: delta = h1 @ fc_W^T + fc_b  (96 small dots, h1 read as bf16)
        if (tid < MT * 3) {
            const int m = tid / 3, j = tid % 3;
            float d = p.fc_b[j];
            const float* w = p.fc_W + j * H;
#pragma unroll 4
            for (int k = 0; k < H; ++k) d += bf2f(abuf[m * AS + 256 + k]) * w[k];
            dbuf[m * 4 + j] = d;
        }
        __syncthreads();
        // pos update + renormalize + emit
        if (tid < MT) {
            const int m = tid;
            float nx = xpos[m * 4 + 0] + dbuf[m * 4 + 0];
            float ny = xpos[m * 4 + 1] + dbuf[m * 4 + 1];
            float nz = xpos[m * 4 + 2] + dbuf[m * 4 + 2];
            const float inv = rsqrtf(nx * nx + ny * ny + nz * nz);
            nx *= inv; ny *= inv; nz *= inv;
            xpos[m * 4 + 0] = nx; xpos[m * 4 + 1] = ny; xpos[m * 4 + 2] = nz;
            abuf[m * AS + 512 + 0] = f2bf(nx);
            abuf[m * AS + 512 + 1] = f2bf(ny);
            abuf[m * AS + 512 + 2] = f2bf(nz);
            float* o = p.out + (b0 + m) * (HW * 3) + s * 3;
            o[0] = nx; o[1] = ny; o[2] = nz;
        }
        __syncthreads();
    }
}

// ---------------------------------------------------------------------------
extern "C" void kernel_launch(void* const* d_in, const int* in_sizes, int n_in,
                              void* d_out, int out_size, void* d_ws, size_t ws_size,
                              hipStream_t stream) {
    (void)in_sizes; (void)n_in; (void)out_size; (void)ws_size;

    unsigned short* wq = (unsigned short*)d_ws;
    float* bias = (float*)((char*)d_ws + (size_t)WQ_TOTAL * sizeof(unsigned short));

    ConvParams cp;
    cp.Whh0[0] = (const float*)d_in[3];  cp.Whh0[1] = (const float*)d_in[11];
    cp.Wih0[0] = (const float*)d_in[2];  cp.Wih0[1] = (const float*)d_in[10];
    cp.Wih1[0] = (const float*)d_in[6];  cp.Wih1[1] = (const float*)d_in[14];
    cp.Whh1[0] = (const float*)d_in[7];  cp.Whh1[1] = (const float*)d_in[15];
    cp.bih0[0] = (const float*)d_in[4];  cp.bhh0[0] = (const float*)d_in[5];
    cp.bih1[0] = (const float*)d_in[8];  cp.bhh1[0] = (const float*)d_in[9];
    cp.bih0[1] = (const float*)d_in[12]; cp.bhh0[1] = (const float*)d_in[13];
    cp.bih1[1] = (const float*)d_in[16]; cp.bhh1[1] = (const float*)d_in[17];
    cp.wq = wq; cp.bias = bias;
    const int conv_items = WQ_TOTAL + 4 * NG;
    pack_weights_kernel<<<(conv_items + 255) / 256, 256, 0, stream>>>(cp);

    Params p;
    p.enc_in = (const float*)d_in[0];
    p.dec_in = (const float*)d_in[1];
    p.fc_W   = (const float*)d_in[18];
    p.fc_b   = (const float*)d_in[19];
    p.wq     = wq;
    p.bias   = bias;
    p.out    = (float*)d_out;
    lstm_seq_kernel<<<B_TOTAL / MT, 256, 0, stream>>>(p);
}
